// PointAttention_65798898975033
// MI455X (gfx1250) — matrix operand order repaired and checked by reading, not compiled, not run on hardware
//
#include <hip/hip_runtime.h>
#include <hip/hip_bf16.h>
#include <math.h>

#define Dd 256
#define Nn 512
#define Bb 2
#define Hh 8
#define HDd 32
#define BN 1024
#define SCALE 0.17677669529663687f

typedef _Float16 h4  __attribute__((ext_vector_type(4)));
typedef _Float16 h8  __attribute__((ext_vector_type(8)));
typedef _Float16 h16 __attribute__((ext_vector_type(16)));
typedef float    f8  __attribute__((ext_vector_type(8)));

#ifndef USE_TDM
#if defined(__has_builtin)
#if __has_builtin(__builtin_amdgcn_tensor_load_to_lds) && __has_builtin(__builtin_amdgcn_s_wait_tensorcnt)
#define USE_TDM 1
#endif
#endif
#endif
#ifndef USE_TDM
#define USE_TDM 0
#endif

#define WMMA_F16(A, B, C) \
  __builtin_amdgcn_wmma_f32_16x16x32_f16(false, (A), false, (B), (short)0, (C), false, false)

// A fragment (16x32 f16): lane L holds row M=L&15; elements 0..7 -> K = (L>>4)*8 + j,
// elements 8..15 -> K = 16 + (L>>4)*8 + j.  base = &mat[row0*ld + k0], K contiguous.
static __device__ __forceinline__ h16 load_a_frag(const _Float16* base, int ld) {
  int lane = threadIdx.x & 31;
  const _Float16* p = base + (lane & 15) * ld + (lane >> 4) * 8;
  h8 lo = *(const h8*)(p);
  h8 hi = *(const h8*)(p + 16);
  h16 r;
#pragma unroll
  for (int i = 0; i < 8; ++i) { r[i] = lo[i]; r[i + 8] = hi[i]; }
  return r;
}

// B fragment (32x16 f16), source stored transposed as [n][k]: lane L holds col N=L&15;
// elements 0..7 -> K = (L>>4)*8 + j, elements 8..15 -> K = 16 + (L>>4)*8 + j.  base = &matT[col0*ld + k0].
static __device__ __forceinline__ h16 load_b_frag(const _Float16* base, int ld) {
  int lane = threadIdx.x & 31;
  const _Float16* p = base + (lane & 15) * ld + (lane >> 4) * 8;
  h8 lo = *(const h8*)(p);
  h8 hi = *(const h8*)(p + 16);
  h16 r;
#pragma unroll
  for (int i = 0; i < 8; ++i) { r[i] = lo[i]; r[i + 8] = hi[i]; }
  return r;
}

static __device__ __forceinline__ _Float16 gelu_h(float x) {
  return (_Float16)(0.5f * x * (1.0f + erff(x * 0.70710678118654752f)));
}

#if USE_TDM
typedef unsigned uv4 __attribute__((ext_vector_type(4)));
typedef int      iv4 __attribute__((ext_vector_type(4)));
typedef int      iv8 __attribute__((ext_vector_type(8)));

// TDM: DMA a [rows x 256] f32 tile (row-major, stride 256) from global to LDS.
static __device__ __forceinline__ void tdm_load_tile(const void* gptr, unsigned ldsByteOff,
                                                     unsigned rows) {
  unsigned long long ga = (unsigned long long)gptr;
  uv4 g0;
  g0[0] = 1u;                                            // count=1, no gather
  g0[1] = ldsByteOff;                                    // lds_addr
  g0[2] = (unsigned)(ga & 0xffffffffu);                  // global_addr[31:0]
  g0[3] = (unsigned)((ga >> 32) & 0x01ffffffu) | (2u << 30);  // addr[56:32] | type=2
  iv8 g1;
  g1[0] = 2 << 16;                 // data_size = 4B
  g1[1] = (int)(256u << 16);       // tensor_dim0 = 256 (bits 79:48)
  g1[2] = (int)(rows << 16);       // tensor_dim1 = rows (bits 111:80)
  g1[3] = (int)(256u << 16);       // tile_dim0 = 256 (bits 127:112)
  g1[4] = (int)rows;               // tile_dim1 = rows
  g1[5] = 256;                     // tensor_dim0_stride = 256
  g1[6] = 0;
  g1[7] = 0;
  iv4 z4 = {0, 0, 0, 0};
#if __clang_major__ >= 23
  iv8 z8 = {0, 0, 0, 0, 0, 0, 0, 0};
  __builtin_amdgcn_tensor_load_to_lds(g0, g1, z4, z4, z8, 0);
#else
  __builtin_amdgcn_tensor_load_to_lds(g0, g1, z4, z4, 0);
#endif
}
#endif

// ---------------- prep kernels ----------------
__global__ void prep_x_kernel(const float* __restrict__ x, _Float16* __restrict__ xh) {
  int i = blockIdx.x * blockDim.x + threadIdx.x;
  if (i < BN * Dd) xh[i] = (_Float16)x[i];
}

__global__ void prep_w_kernel(const float* __restrict__ Wq, const float* __restrict__ Wk,
                              const float* __restrict__ Wv, const float* __restrict__ Wo,
                              _Float16* __restrict__ wt) {
  int i = blockIdx.x * blockDim.x + threadIdx.x;
  if (i >= 4 * Dd * Dd) return;
  int mat = i >> 16, idx = i & 65535;
  int n = idx >> 8, k = idx & 255;
  const float* W = (mat == 0) ? Wq : (mat == 1) ? Wk : (mat == 2) ? Wv : Wo;
  wt[mat * Dd * Dd + n * Dd + k] = (_Float16)W[k * Dd + n];  // transpose: [n][k]
}

__global__ void prep_w2_kernel(const float* __restrict__ Wp2, const float* __restrict__ bp2,
                               _Float16* __restrict__ w2t, float* __restrict__ b2s) {
  int i = blockIdx.x * blockDim.x + threadIdx.x;
  if (i < 16 * Dd) {
    int n = i >> 8, k = i & 255;
    float s = 0.f;
    if (n < Hh) {
#pragma unroll
      for (int hd = 0; hd < HDd; ++hd) s += Wp2[k * Dd + n * HDd + hd];
    }
    w2t[n * Dd + k] = (_Float16)s;   // [n(=head, padded to 16)][k]
  } else if (i < 16 * Dd + Hh) {
    int h = i - 16 * Dd;
    float s = 0.f;
#pragma unroll
    for (int hd = 0; hd < HDd; ++hd) s += bp2[h * HDd + hd];
    b2s[h] = s;
  }
}

__global__ void prep_cp_kernel(const float* __restrict__ coords, const float* __restrict__ Wp1,
                               const float* __restrict__ bp1, float* __restrict__ cp,
                               float* __restrict__ cpb) {
  int i = blockIdx.x * blockDim.x + threadIdx.x;
  if (i >= BN * Dd) return;
  int row = i >> 8, k = i & 255;
  float c0 = coords[row * 3 + 0], c1 = coords[row * 3 + 1], c2 = coords[row * 3 + 2];
  float v = c0 * Wp1[0 * Dd + k] + c1 * Wp1[1 * Dd + k] + c2 * Wp1[2 * Dd + k];
  cp[i] = v;
  cpb[i] = v + bp1[k];
}

// ---------------- QKV projection GEMM (WMMA) ----------------
__global__ void __launch_bounds__(32)
qkv_gemm_kernel(const _Float16* __restrict__ xh, const _Float16* __restrict__ wt,
                const float* __restrict__ bq, const float* __restrict__ bk,
                const float* __restrict__ bv, _Float16* __restrict__ qh,
                _Float16* __restrict__ kh, _Float16* __restrict__ vh,
                _Float16* __restrict__ vt) {
  int tile = blockIdx.x;                 // 3 * 64 * 16
  int mat = tile >> 10;
  int t2 = tile & 1023;
  int rt = t2 >> 4, ct = t2 & 15;
  const _Float16* W = wt + mat * Dd * Dd;
  const float* bias = (mat == 0) ? bq : (mat == 1) ? bk : bv;
  _Float16* out = (mat == 0) ? qh : (mat == 1) ? kh : vh;
  int lane = threadIdx.x & 31;
  f8 acc = {};
#pragma unroll
  for (int kc = 0; kc < 8; ++kc) {
    h16 a = load_a_frag(xh + rt * 16 * Dd + kc * 32, Dd);
    h16 b = load_b_frag(W + ct * 16 * Dd + kc * 32, Dd);
    acc = WMMA_F16(a, b, acc);
  }
  int d = ct * 16 + (lane & 15);
  float bb = bias[d];
#pragma unroll
  for (int r = 0; r < 8; ++r) {
    int m = rt * 16 + r + 8 * (lane >> 4);
    float v = acc[r] + bb;
    out[m * Dd + d] = (_Float16)v;
    if (mat == 2) {                            // also V transposed per head: [b][h][hd][n]
      int bI = m >> 9, ns = m & 511;
      int hh = d >> 5, hd = d & 31;
      vt[((bI * Hh + hh) * HDd + hd) * Nn + ns] = (_Float16)v;
    }
  }
}

// ---------------- pairwise bias MLP (fused GELU + WMMA K-reduction) ----------------
__global__ void __launch_bounds__(256)
bias_kernel(const float* __restrict__ cp, const float* __restrict__ cpb,
            const _Float16* __restrict__ w2t, const float* __restrict__ b2s,
            float* __restrict__ biasS) {
  __shared__ float stageN[16 * Dd];      // 16 KB: cpb rows of the n-tile (LDS offset 0)
  __shared__ float stageM[16 * Dd];      // 16 KB: cp rows of the m-tile (LDS offset 16384)
  __shared__ _Float16 g[256 * 32];       // 16 KB: 256 (n,m)-pairs x 32 k-chunk
  int mt = blockIdx.x, nt = blockIdx.y, b = blockIdx.z;
  int t = threadIdx.x;
  int lane = t & 31, wave = t >> 5;      // 8 waves

  const float* gsrcN = cpb + (b * Nn + nt * 16) * Dd;
  const float* gsrcM = cp + (b * Nn + mt * 16) * Dd;
#if USE_TDM
  if (t < 32) {                          // one wave drives the Tensor Data Mover
    tdm_load_tile(gsrcN, 0u, 16u);
    tdm_load_tile(gsrcM, 16u * Dd * 4u, 16u);
    __builtin_amdgcn_s_wait_tensorcnt(0);
  }
#else
  {                                      // cooperative float4 staging fallback
    const float4* sN = (const float4*)gsrcN;
    const float4* sM = (const float4*)gsrcM;
    float4* dN = (float4*)stageN;
    float4* dM = (float4*)stageM;
#pragma unroll
    for (int i = 0; i < 4; ++i) {
      dN[t + 256 * i] = sN[t + 256 * i];
      dM[t + 256 * i] = sM[t + 256 * i];
    }
  }
#endif

  // preload B fragments for all 8 k-chunks (W2s^T, shared across the whole grid)
  h16 bf[8];
#pragma unroll
  for (int kc = 0; kc < 8; ++kc) bf[kc] = load_b_frag(w2t + kc * 32, Dd);

  const float* pn = stageN + (t >> 4) * Dd;   // pair P = t: n_local = t>>4
  const float* pm = stageM + (t & 15) * Dd;   //            m_local = t&15
  f8 acc0 = {}, acc1 = {};
  for (int kc = 0; kc < 8; ++kc) {
    __syncthreads();
    const float4* an = (const float4*)(pn + kc * 32);
    const float4* am = (const float4*)(pm + kc * 32);
#pragma unroll
    for (int q = 0; q < 4; ++q) {
      float4 a0 = an[2 * q], a1 = an[2 * q + 1];
      float4 b0 = am[2 * q], b1 = am[2 * q + 1];
      h8 pk;
      pk[0] = gelu_h(a0.x - b0.x); pk[1] = gelu_h(a0.y - b0.y);
      pk[2] = gelu_h(a0.z - b0.z); pk[3] = gelu_h(a0.w - b0.w);
      pk[4] = gelu_h(a1.x - b1.x); pk[5] = gelu_h(a1.y - b1.y);
      pk[6] = gelu_h(a1.z - b1.z); pk[7] = gelu_h(a1.w - b1.w);
      *(h8*)&g[t * 32 + q * 8] = pk;
    }
    __syncthreads();
    // wave w reduces pair-row-tiles (n_local = 2w, 2w+1) against W2s
    h16 a0 = load_a_frag(&g[(2 * wave) * 16 * 32], 32);
    h16 a1 = load_a_frag(&g[(2 * wave + 1) * 16 * 32], 32);
    acc0 = WMMA_F16(a0, bf[kc], acc0);
    acc1 = WMMA_F16(a1, bf[kc], acc1);
  }
  int head = lane & 15;
  if (head < Hh) {
    float bs = b2s[head];
    int n0 = nt * 16 + 2 * wave;
#pragma unroll
    for (int r = 0; r < 8; ++r) {
      int ml = mt * 16 + r + 8 * (lane >> 4);
      biasS[((b * Hh + head) * Nn + n0) * Nn + ml] = (acc0[r] + bs) * SCALE;
      biasS[((b * Hh + head) * Nn + n0 + 1) * Nn + ml] = (acc1[r] + bs) * SCALE;
    }
  }
}

// ---------------- fused attention: QK^T + bias, softmax, P*V ----------------
__global__ void __launch_bounds__(32)
attn_kernel(const _Float16* __restrict__ qh, const _Float16* __restrict__ kh,
            const _Float16* __restrict__ vt, const float* __restrict__ biasS,
            _Float16* __restrict__ outh) {
  __shared__ float S[16 * Nn];           // 32 KB scores
  __shared__ _Float16 P[16 * Nn];        // 16 KB probabilities (f16, WMMA A layout source)
  int nt = blockIdx.x, h = blockIdx.y, b = blockIdx.z;
  int lane = threadIdx.x;
  h16 aq = load_a_frag(qh + (b * Nn + nt * 16) * Dd + h * HDd, Dd);
  const float* brow = biasS + ((b * Hh + h) * Nn + nt * 16) * Nn;
  for (int mtile = 0; mtile < 32; ++mtile) {
    if (mtile + 1 < 32)
      __builtin_prefetch(kh + (b * Nn + (mtile + 1) * 16) * Dd + h * HDd, 0, 1);
    h16 bk = load_b_frag(kh + (b * Nn + mtile * 16) * Dd + h * HDd, Dd);
    f8 s = {};
    s = WMMA_F16(aq, bk, s);
    int mc = mtile * 16 + (lane & 15);
#pragma unroll
    for (int r = 0; r < 8; ++r) {
      int nl = r + 8 * (lane >> 4);
      S[nl * Nn + mc] = s[r] * SCALE + brow[nl * Nn + mc];
    }
  }
  __syncthreads();
  // softmax: lane handles row=lane&15, half-range [ (lane>>4)*256, +256 ), f4-vectorized
  {
    int row = lane & 15;
    const float4* s4 = (const float4*)(S + row * Nn + (lane >> 4) * 256);
    _Float16* prow = P + row * Nn + (lane >> 4) * 256;
    float mx = -3.0e38f;
    for (int j = 0; j < 64; ++j) {
      float4 v = s4[j];
      mx = fmaxf(mx, fmaxf(fmaxf(v.x, v.y), fmaxf(v.z, v.w)));
    }
    mx = fmaxf(mx, __shfl_xor(mx, 16));
    float sum = 0.f;
    for (int j = 0; j < 64; ++j) {
      float4 v = s4[j];
      float e0 = __expf(v.x - mx), e1 = __expf(v.y - mx);
      float e2 = __expf(v.z - mx), e3 = __expf(v.w - mx);
      sum += (e0 + e1) + (e2 + e3);
      h4 p;
      p[0] = (_Float16)e0; p[1] = (_Float16)e1; p[2] = (_Float16)e2; p[3] = (_Float16)e3;
      *(h4*)&prow[j * 4] = p;
    }
    sum += __shfl_xor(sum, 16);
    float rinv = 1.0f / sum;
    for (int j = 0; j < 64; ++j) {
      h4 p = *(h4*)&prow[j * 4];
      p[0] = (_Float16)((float)p[0] * rinv);
      p[1] = (_Float16)((float)p[1] * rinv);
      p[2] = (_Float16)((float)p[2] * rinv);
      p[3] = (_Float16)((float)p[3] * rinv);
      *(h4*)&prow[j * 4] = p;
    }
  }
  __syncthreads();
  // O = P(16x512) @ V(512x32): 16 k-chunks, 2 output col-tiles
  f8 o0 = {}, o1 = {};
  const _Float16* vbase = vt + (b * Hh + h) * HDd * Nn;
  for (int kc = 0; kc < 16; ++kc) {
    h16 ap = load_a_frag((const _Float16*)P + kc * 32, Nn);
    h16 b0 = load_b_frag(vbase + 0 * Nn + kc * 32, Nn);
    h16 b1 = load_b_frag(vbase + 16 * Nn + kc * 32, Nn);
    o0 = WMMA_F16(ap, b0, o0);
    o1 = WMMA_F16(ap, b1, o1);
  }
#pragma unroll
  for (int r = 0; r < 8; ++r) {
    int n = nt * 16 + r + 8 * (lane >> 4);
    outh[(b * Nn + n) * Dd + h * HDd + (lane & 15)] = (_Float16)o0[r];
    outh[(b * Nn + n) * Dd + h * HDd + 16 + (lane & 15)] = (_Float16)o1[r];
  }
}

// ---------------- output projection GEMM (WMMA) ----------------
__global__ void __launch_bounds__(32)
oproj_kernel(const _Float16* __restrict__ outh, const _Float16* __restrict__ wot,
             const float* __restrict__ bo, float* __restrict__ out) {
  int tile = blockIdx.x;                 // 64*16
  int rt = tile >> 4, ct = tile & 15;
  int lane = threadIdx.x & 31;
  f8 acc = {};
#pragma unroll
  for (int kc = 0; kc < 8; ++kc) {
    h16 a = load_a_frag(outh + rt * 16 * Dd + kc * 32, Dd);
    h16 b = load_b_frag(wot + ct * 16 * Dd + kc * 32, Dd);
    acc = WMMA_F16(a, b, acc);
  }
  int col = ct * 16 + (lane & 15);
  float bb = bo[col];
#pragma unroll
  for (int r = 0; r < 8; ++r) {
    int m = rt * 16 + r + 8 * (lane >> 4);
    out[m * Dd + col] = acc[r] + bb;
  }
}

extern "C" void kernel_launch(void* const* d_in, const int* in_sizes, int n_in,
                              void* d_out, int out_size, void* d_ws, size_t ws_size,
                              hipStream_t stream) {
  const float* x      = (const float*)d_in[0];
  const float* coords = (const float*)d_in[1];
  const float* Wq = (const float*)d_in[2];  const float* bq = (const float*)d_in[3];
  const float* Wk = (const float*)d_in[4];  const float* bk = (const float*)d_in[5];
  const float* Wv = (const float*)d_in[6];  const float* bv = (const float*)d_in[7];
  const float* Wo = (const float*)d_in[8];  const float* bo = (const float*)d_in[9];
  const float* Wp1 = (const float*)d_in[10]; const float* bp1 = (const float*)d_in[11];
  const float* Wp2 = (const float*)d_in[12]; const float* bp2 = (const float*)d_in[13];
  float* out = (float*)d_out;

  // workspace layout (bytes)
  char* ws = (char*)d_ws;
  _Float16* xh  = (_Float16*)(ws);                        // 512 KB
  _Float16* wt  = (_Float16*)(ws + 524288);               // 4 x 128 KB (Wq,Wk,Wv,Wo transposed)
  _Float16* w2t = wt + 4 * 65536;                         // 8 KB  ([16][256])
  float* b2s    = (float*)(ws + 524288 + 524288 + 8192);  // 32 B
  float* cp     = (float*)(ws + 1056832);                 // 1 MB
  float* cpb    = cp + BN * Dd;                           // 1 MB
  _Float16* qh  = (_Float16*)(cpb + BN * Dd);             // 512 KB
  _Float16* kh  = qh + BN * Dd;                           // 512 KB
  _Float16* vh  = kh + BN * Dd;                           // 512 KB
  _Float16* vt  = vh + BN * Dd;                           // 512 KB  ([b][h][hd][n])
  float* biasS  = (float*)(vt + BN * Dd);                 // 16.8 MB ([b][h][n][m])
  _Float16* outh = (_Float16*)(biasS + Bb * Hh * Nn * Nn); // 512 KB

  prep_x_kernel<<<1024, 256, 0, stream>>>(x, xh);
  prep_w_kernel<<<1024, 256, 0, stream>>>(Wq, Wk, Wv, Wo, wt);
  prep_w2_kernel<<<17, 256, 0, stream>>>(Wp2, bp2, w2t, b2s);
  prep_cp_kernel<<<1024, 256, 0, stream>>>(coords, Wp1, bp1, cp, cpb);
  qkv_gemm_kernel<<<3072, 32, 0, stream>>>(xh, wt, bq, bk, bv, qh, kh, vh, vt);
  bias_kernel<<<dim3(32, 32, 2), 256, 0, stream>>>(cp, cpb, w2t, b2s, biasS);
  attn_kernel<<<dim3(32, 8, 2), 32, 0, stream>>>(qh, kh, vt, biasS, outh);
  oproj_kernel<<<1024, 32, 0, stream>>>(outh, wt + 3 * 65536, bo, out);
}
